// TransformerDecoderV2_34359738850
// MI455X (gfx1250) — matrix operand order repaired and checked by reading, not compiled, not run on hardware
//
#include <hip/hip_runtime.h>

// ---------------------------------------------------------------------------
// Problem constants (from reference): B=256,S=64,D=512,H=8,L=6,DFF=2048,
// LAT=256, V=16, N=64.  B*S = 16384 tokens.
// ---------------------------------------------------------------------------
#define BB   256
#define SS   64
#define DD   512
#define HH   8
#define LL   6
#define DFF  2048
#define LAT  256
#define VV   16
#define NN   64
#define MTOK (BB * SS)   // 16384

typedef unsigned short u16;
typedef unsigned int   u32;

typedef __attribute__((ext_vector_type(16))) __bf16 v16bf;
typedef __attribute__((ext_vector_type(8)))  float  v8f;

__device__ __forceinline__ u16 f2bf(float f) {
    u32 x = __float_as_uint(f);
    u32 r = x + 0x7FFFu + ((x >> 16) & 1u);   // round-to-nearest-even
    return (u16)(r >> 16);
}
__device__ __forceinline__ float bf2f(u16 h) {
    return __uint_as_float(((u32)h) << 16);
}

// gfx1250 async global->LDS DMA (ASYNCcnt-tracked), ISA §15.18.3 op 98.
// VDST = per-lane LDS byte address, VADDR = per-lane 64-bit global address.
__device__ __forceinline__ void async_copy_b128(u32 lds_addr, const void* gaddr) {
    asm volatile("global_load_async_to_lds_b128 %0, %1, off"
                 :: "v"(lds_addr), "v"(gaddr) : "memory");
}
__device__ __forceinline__ void wait_async_le8() {
    asm volatile("s_wait_asynccnt 8" ::: "memory");
}
__device__ __forceinline__ void wait_async_0() {
    asm volatile("s_wait_asynccnt 0" ::: "memory");
}

// ---------------------------------------------------------------------------
// bf16 WMMA GEMM:  C[M,N] = act(A[M,K] @ W[N,K]^T + bias)
//   Block = 128 threads = 4 waves (2x2), block tile 128x128, K-step 32.
//   Each wave owns a 64x64 quadrant = 4x4 fragments -> 16
//   v_wmma_f32_16x16x32_bf16 per K-step against 8 fragment loads.
//   Double-buffered LDS; full-width N tiles stream in via
//   global_load_async_to_lds_b128 (8 per-wave ops per tile, s_wait_asynccnt),
//   partial N tiles use guarded loads with zero fill.
//   M multiple of 128, K multiple of 32 (true for all calls); N checked.
//   flags bit0 = ReLU.  Cf (fp32) and/or Cb (bf16) outputs, ldc = N.
// ---------------------------------------------------------------------------
#define GTM 128
#define GTN 128
#define GTK 32
#define LDP 48   // padded LDS row stride in u16 (96 B: 16-B aligned, bank-spread)

__global__ void __launch_bounds__(128)
gemm_bf16_kernel(const u16* __restrict__ A, const u16* __restrict__ W,
                 const float* __restrict__ bias, float* __restrict__ Cf,
                 u16* __restrict__ Cb, int M, int N, int K, int flags)
{
    __shared__ u16 AsT[2][GTM][LDP];
    __shared__ u16 WsT[2][GTN][LDP];

    const int tid  = threadIdx.x;
    const int wave = tid >> 5;
    const int lane = tid & 31;
    const int wr   = wave >> 1;     // wave row (0..1) -> 64 rows
    const int wc   = wave & 1;      // wave col (0..1) -> 64 cols
    const int gm   = blockIdx.y * GTM;
    const int gn   = blockIdx.x * GTN;
    const int half = lane >> 4;     // 0: lanes 0-15, 1: lanes 16-31
    const int l16  = lane & 15;

    const v8f vzero = {0.f, 0.f, 0.f, 0.f, 0.f, 0.f, 0.f, 0.f};
    v8f acc[4][4];
    #pragma unroll
    for (int i = 0; i < 4; ++i)
        #pragma unroll
        for (int j = 0; j < 4; ++j) acc[i][j] = vzero;

    union Frag { v16bf v; u16 u[16]; };

    const bool fullN = (gn + GTN) <= N;
    const int  nk    = K / GTK;

    // ---- tile staging -------------------------------------------------------
    // 128 rows x 32 cols per matrix = 512 16-byte vectors; 4 per thread.
    auto stage_async = [&](int k0, int b) {
        #pragma unroll
        for (int it = 0; it < 4; ++it) {
            int idx = tid + it * 128;
            int r   = idx >> 2;               // 0..127
            int c   = (idx & 3) << 3;         // 0,8,16,24
            async_copy_b128((u32)(size_t)&AsT[b][r][c],
                            A + (size_t)(gm + r) * K + k0 + c);
            async_copy_b128((u32)(size_t)&WsT[b][r][c],
                            W + (size_t)(gn + r) * K + k0 + c);
        }
    };
    auto stage_sync = [&](int k0, int b) {
        #pragma unroll
        for (int it = 0; it < 4; ++it) {
            int idx = tid + it * 128;
            int r   = idx >> 2;
            int c   = (idx & 3) << 3;
            const uint4 va = *(const uint4*)(A + (size_t)(gm + r) * K + k0 + c);
            *(uint4*)&AsT[b][r][c] = va;
            uint4 vw = make_uint4(0u, 0u, 0u, 0u);
            int wn = gn + r;
            if (wn < N) vw = *(const uint4*)(W + (size_t)wn * K + k0 + c);
            *(uint4*)&WsT[b][r][c] = vw;
        }
        if (k0 + GTK < K)   // -> global_prefetch_b8
            __builtin_prefetch(A + (size_t)(gm + (tid >> 1)) * K + k0 + GTK, 0, 0);
    };

    if (fullN) stage_async(0, 0); else stage_sync(0, 0);

    for (int kt = 0; kt < nk; ++kt) {
        const int cur = kt & 1;
        if (fullN) {
            if (kt + 1 < nk) { stage_async((kt + 1) * GTK, cur ^ 1); wait_async_le8(); }
            else             { wait_async_0(); }
        } else {
            if (kt + 1 < nk) stage_sync((kt + 1) * GTK, cur ^ 1);
        }
        __syncthreads();

        // ---- fragments per ISA 16-bit A (16x32) / B (32x16) VGPR layouts ----
        Frag a[4], b[4];
        #pragma unroll
        for (int mi = 0; mi < 4; ++mi) {
            int mrow = wr * 64 + mi * 16 + l16;
            #pragma unroll
            for (int v = 0; v < 8; ++v) {
                int kk = (v < 4 ? (v * 2) : (16 + (v - 4) * 2)) + half * 8;
                a[mi].u[2 * v]     = AsT[cur][mrow][kk];
                a[mi].u[2 * v + 1] = AsT[cur][mrow][kk + 1];
            }
        }
        #pragma unroll
        for (int ni = 0; ni < 4; ++ni) {
            int nrow = wc * 64 + ni * 16 + l16;
            #pragma unroll
            for (int v = 0; v < 8; ++v) {
                int kk = (v < 4 ? (v * 2) : (16 + (v - 4) * 2)) + half * 8;
                b[ni].u[2 * v]     = WsT[cur][nrow][kk];
                b[ni].u[2 * v + 1] = WsT[cur][nrow][kk + 1];
            }
        }

        // ---- 16 WMMAs per wave per K-step ----
        #pragma unroll
        for (int mi = 0; mi < 4; ++mi)
            #pragma unroll
            for (int ni = 0; ni < 4; ++ni)
                acc[mi][ni] = __builtin_amdgcn_wmma_f32_16x16x32_bf16(
                    false, a[mi].v, false, b[ni].v,
                    (short)0, acc[mi][ni], false, false);
        __syncthreads();
    }

    // ---- epilogue: bias / ReLU / fp32 + bf16 stores -------------------------
    #pragma unroll
    for (int mi = 0; mi < 4; ++mi) {
        #pragma unroll
        for (int ni = 0; ni < 4; ++ni) {
            int col = gn + wc * 64 + ni * 16 + l16;
            if (col < N) {
                float bv = bias ? bias[col] : 0.0f;
                #pragma unroll
                for (int v = 0; v < 8; ++v) {
                    int rowg = gm + wr * 64 + mi * 16 + half * 8 + v;
                    float val = acc[mi][ni][v] + bv;
                    if (flags & 1) val = fmaxf(val, 0.0f);
                    size_t o = (size_t)rowg * N + col;
                    if (Cf) Cf[o] = val;
                    if (Cb) Cb[o] = f2bf(val);
                }
            }
        }
    }
}

// ---------------------------------------------------------------------------
// fp32 -> bf16 converter (weights, latent)
// ---------------------------------------------------------------------------
__global__ void __launch_bounds__(256)
f32_to_bf16_kernel(const float* __restrict__ src, u16* __restrict__ dst, int n)
{
    int i = blockIdx.x * 256 + threadIdx.x;
    if (i < n) dst[i] = f2bf(src[i]);
}

// ---------------------------------------------------------------------------
// Positional encoding table  pe[S,D]
// ---------------------------------------------------------------------------
__global__ void __launch_bounds__(256)
pe_kernel(float* __restrict__ pe)
{
    int i = blockIdx.x * 256 + threadIdx.x;
    if (i >= SS * DD) return;
    int s = i >> 9, d = i & (DD - 1);
    float freq = __expf((float)(d & ~1) * (-9.210340371976184f / (float)DD));
    float a = (float)s * freq;
    pe[i] = (d & 1) ? __cosf(a) : __sinf(a);
}

// ---------------------------------------------------------------------------
// Embedding gather + value linear -> bf16 emb row [512]
// ---------------------------------------------------------------------------
__global__ void __launch_bounds__(128)
embed_kernel(const float* __restrict__ teacher, const float* __restrict__ te,
             const float* __restrict__ nae, const float* __restrict__ nbe,
             const float* __restrict__ vw, const float* __restrict__ vb,
             u16* __restrict__ emb)
{
    int row = blockIdx.x;
    int t   = threadIdx.x;
    const float* ts = teacher + (size_t)row * 4;
    int ct = (int)ts[0]; ct = ct < 0 ? 0 : (ct > VV - 1 ? VV - 1 : ct);
    int na = (int)ts[1]; na = na < 0 ? 0 : (na > NN - 1 ? NN - 1 : na);
    int nb = (int)ts[2]; nb = nb < 0 ? 0 : (nb > NN - 1 ? NN - 1 : nb);
    float val = ts[3];
    u16* o = emb + (size_t)row * DD;
    o[t]       = f2bf(te[ct * 128 + t]);
    o[128 + t] = f2bf(nae[na * 128 + t]);
    o[256 + t] = f2bf(nbe[nb * 128 + t]);
    o[384 + t] = f2bf(val * vw[t] + vb[t]);
}

// ---------------------------------------------------------------------------
// x = proj_out + pe[row % S]  (writes fp32 + bf16)
// ---------------------------------------------------------------------------
__global__ void __launch_bounds__(256)
addpe_kernel(const float* __restrict__ gin, const float* __restrict__ pe,
             float* __restrict__ xf, u16* __restrict__ xb)
{
    int i = blockIdx.x * 256 + threadIdx.x;
    int s = (i >> 9) & (SS - 1);
    int d = i & (DD - 1);
    float v = gin[i] + pe[(s << 9) + d];
    xf[i] = v;
    xb[i] = f2bf(v);
}

// ---------------------------------------------------------------------------
// Fused residual + LayerNorm over D=512.  bmode: 0 delta per row, 1 per batch.
// ---------------------------------------------------------------------------
__global__ void __launch_bounds__(256)
ln_kernel(float* __restrict__ x, const float* __restrict__ delta, int bmode,
          const float* __restrict__ g, const float* __restrict__ b,
          u16* __restrict__ xb)
{
    int row = blockIdx.x, t = threadIdx.x;
    __shared__ float red[256];
    size_t xo = (size_t)row * DD;
    const float* dp = delta + (bmode ? (size_t)(row >> 6) * DD : xo);
    float v0 = x[xo + t]       + dp[t];
    float v1 = x[xo + 256 + t] + dp[256 + t];
    red[t] = v0 + v1;
    __syncthreads();
    for (int st = 128; st > 0; st >>= 1) { if (t < st) red[t] += red[t + st]; __syncthreads(); }
    float mean = red[0] * (1.0f / (float)DD);
    __syncthreads();
    float d0 = v0 - mean, d1 = v1 - mean;
    red[t] = d0 * d0 + d1 * d1;
    __syncthreads();
    for (int st = 128; st > 0; st >>= 1) { if (t < st) red[t] += red[t + st]; __syncthreads(); }
    float inv = rsqrtf(red[0] * (1.0f / (float)DD) + 1e-5f);
    float o0 = d0 * inv * g[t]       + b[t];
    float o1 = d1 * inv * g[256 + t] + b[256 + t];
    x[xo + t] = o0;        x[xo + 256 + t] = o1;
    xb[xo + t] = f2bf(o0); xb[xo + 256 + t] = f2bf(o1);
}

// ---------------------------------------------------------------------------
// Causal self-attention for one (batch, head): S=64, hd=64.
// ---------------------------------------------------------------------------
__global__ void __launch_bounds__(64)
attn_kernel(const u16* __restrict__ qkv, u16* __restrict__ att)
{
    int bh = blockIdx.x;
    int b  = bh / HH, h = bh % HH;
    __shared__ float Kl[64][65];
    __shared__ float Vl[64][65];
    __shared__ float Sc[64][65];
    int t = threadIdx.x;                     // query row
    const u16* base = qkv + (size_t)(b * SS + t) * (3 * DD) + h * 64;
    float q[64];
    #pragma unroll
    for (int d = 0; d < 64; ++d) {
        q[d]     = bf2f(base[d]);
        Kl[t][d] = bf2f(base[DD + d]);
        Vl[t][d] = bf2f(base[2 * DD + d]);
    }
    __syncthreads();
    float mx = -INFINITY;
    for (int k = 0; k <= t; ++k) {
        float s = 0.f;
        #pragma unroll
        for (int d = 0; d < 64; ++d) s += q[d] * Kl[k][d];
        s *= 0.125f;                         // 1/sqrt(64)
        Sc[t][k] = s;
        mx = fmaxf(mx, s);
    }
    float sum = 0.f;
    for (int k = 0; k <= t; ++k) { float e = __expf(Sc[t][k] - mx); Sc[t][k] = e; sum += e; }
    float inv = 1.0f / sum;
    u16* o = att + (size_t)(b * SS + t) * DD + h * 64;
    for (int d = 0; d < 64; ++d) {
        float accv = 0.f;
        for (int k = 0; k <= t; ++k) accv += Sc[t][k] * Vl[k][d];
        o[d] = f2bf(accv * inv);
    }
}

// ---------------------------------------------------------------------------
// values head: dot(x_row, val_w) + val_b  (N=1)
// ---------------------------------------------------------------------------
__global__ void __launch_bounds__(128)
valhead_kernel(const float* __restrict__ x, const float* __restrict__ w,
               const float* __restrict__ bias, float* __restrict__ out)
{
    int row = blockIdx.x, t = threadIdx.x;
    __shared__ float red[128];
    float s = 0.f;
    for (int c = t; c < DD; c += 128) s += x[(size_t)row * DD + c] * w[c];
    red[t] = s;
    __syncthreads();
    for (int st = 64; st > 0; st >>= 1) { if (t < st) red[t] += red[t + st]; __syncthreads(); }
    if (t == 0) out[row] = red[0] + bias[0];
}

// ---------------------------------------------------------------------------
// node_a sample (argmax of logits + gumbel) + build nb input [x | one_hot]
// ---------------------------------------------------------------------------
__global__ void __launch_bounds__(64)
sample_a_kernel(const float* __restrict__ na_logits, const float* __restrict__ gum,
                const u16* __restrict__ xb, u16* __restrict__ nbin,
                int* __restrict__ naidx)
{
    int row = blockIdx.x, t = threadIdx.x;
    __shared__ int bidx;
    if (t == 0) {
        float best = -INFINITY; int bi = 0;
        for (int n = 0; n < NN; ++n) {
            float v = na_logits[(size_t)row * NN + n] + gum[(size_t)row * NN + n];
            if (v > best) { best = v; bi = n; }
        }
        bidx = bi; naidx[row] = bi;
    }
    __syncthreads();
    for (int c = t; c < DD; c += 64) nbin[(size_t)row * (DD + NN) + c] = xb[(size_t)row * DD + c];
    nbin[(size_t)row * (DD + NN) + DD + t] = (t == bidx) ? (u16)0x3F80 : (u16)0;
}

// ---------------------------------------------------------------------------
// final: type argmax, masked node_b argmax (+gumbel), assemble sequence
// ---------------------------------------------------------------------------
__global__ void __launch_bounds__(256)
finalize_kernel(const float* __restrict__ tl, const float* __restrict__ nbl,
                const float* __restrict__ gum, const int* __restrict__ naidx,
                const float* __restrict__ vals, float* __restrict__ seq)
{
    int row = blockIdx.x * 256 + threadIdx.x;
    if (row >= MTOK) return;
    float best = -INFINITY; int ti = 0;
    for (int v = 0; v < VV; ++v) {
        float x = tl[(size_t)row * VV + v];
        if (x > best) { best = x; ti = v; }
    }
    int na = naidx[row];
    float bb = -INFINITY; int bi = 0;
    for (int n = 0; n < NN; ++n) {
        if (n == na) continue;   // masked to -inf in the reference
        float x = nbl[(size_t)row * NN + n] + gum[(size_t)row * NN + n];
        if (x > bb) { bb = x; bi = n; }
    }
    seq[(size_t)row * 4 + 0] = (float)ti;
    seq[(size_t)row * 4 + 1] = (float)na;
    seq[(size_t)row * 4 + 2] = (float)bi;
    seq[(size_t)row * 4 + 3] = vals[row];
}

// ---------------------------------------------------------------------------
// Host orchestration
// ---------------------------------------------------------------------------
extern "C" void kernel_launch(void* const* d_in, const int* in_sizes, int n_in,
                              void* d_out, int out_size, void* d_ws, size_t ws_size,
                              hipStream_t stream)
{
    (void)in_sizes; (void)n_in; (void)out_size; (void)ws_size;

    const float* latent   = (const float*)d_in[0];
    const float* teacher  = (const float*)d_in[1];
    const float* gumbel_a = (const float*)d_in[2];
    const float* gumbel_b = (const float*)d_in[3];
    const float* type_emb = (const float*)d_in[4];
    const float* na_emb   = (const float*)d_in[5];
    const float* nb_emb   = (const float*)d_in[6];
    const float* value_w  = (const float*)d_in[7];
    const float* value_b  = (const float*)d_in[8];
    const float* proj_w   = (const float*)d_in[9];
    const float* proj_b   = (const float*)d_in[10];
    const float* latent_w = (const float*)d_in[11];
    const float* latent_b = (const float*)d_in[12];
    const float* sa_in_w  = (const float*)d_in[13];
    const float* sa_in_b  = (const float*)d_in[14];
    const float* sa_out_w = (const float*)d_in[15];
    const float* sa_out_b = (const float*)d_in[16];
    // d_in[17], d_in[18] = ca_q_w/b: provably dead (softmax over 1 key == 1)
    const float* ca_kv_w  = (const float*)d_in[19];
    const float* ca_kv_b  = (const float*)d_in[20];
    const float* ca_out_w = (const float*)d_in[21];
    const float* ca_out_b = (const float*)d_in[22];
    const float* ff1_w    = (const float*)d_in[23];
    const float* ff1_b    = (const float*)d_in[24];
    const float* ff2_w    = (const float*)d_in[25];
    const float* ff2_b    = (const float*)d_in[26];
    const float* ln1_g    = (const float*)d_in[27];
    const float* ln1_b    = (const float*)d_in[28];
    const float* ln2_g    = (const float*)d_in[29];
    const float* ln2_b    = (const float*)d_in[30];
    const float* ln3_g    = (const float*)d_in[31];
    const float* ln3_b    = (const float*)d_in[32];
    const float* type_w   = (const float*)d_in[33];
    const float* type_b   = (const float*)d_in[34];
    const float* na_w     = (const float*)d_in[35];
    const float* na_b     = (const float*)d_in[36];
    const float* val_w    = (const float*)d_in[37];
    const float* val_b    = (const float*)d_in[38];
    const float* nb_w     = (const float*)d_in[39];
    const float* nb_b     = (const float*)d_in[40];

    // --- bf16 weight arena offsets (in u16 elements) ---
    constexpr size_t OFF_PROJ   = 0;
    constexpr size_t OFF_LATW   = OFF_PROJ  + (size_t)DD * DD;
    constexpr size_t OFF_SAIN   = OFF_LATW  + (size_t)DD * LAT;
    constexpr size_t OFF_SAOUT  = OFF_SAIN  + (size_t)LL * 3 * DD * DD;
    constexpr size_t OFF_CAKV   = OFF_SAOUT + (size_t)LL * DD * DD;
    constexpr size_t OFF_CAOUT  = OFF_CAKV  + (size_t)LL * 2 * DD * DD;
    constexpr size_t OFF_FF1    = OFF_CAOUT + (size_t)LL * DD * DD;
    constexpr size_t OFF_FF2    = OFF_FF1   + (size_t)LL * DFF * DD;
    constexpr size_t OFF_TYPEW  = OFF_FF2   + (size_t)LL * DD * DFF;
    constexpr size_t OFF_NAW    = OFF_TYPEW + (size_t)VV * DD;
    constexpr size_t OFF_NBW    = OFF_NAW   + (size_t)NN * DD;
    constexpr size_t OFF_LATIN  = OFF_NBW   + (size_t)NN * (DD + NN);
    constexpr size_t OFF_MEMBF  = OFF_LATIN + (size_t)BB * LAT;
    constexpr size_t OFF_MEMVBF = OFF_MEMBF + (size_t)BB * DD;
    constexpr size_t WBF_TOTAL  = OFF_MEMVBF + (size_t)BB * DD;

    char* base = (char*)d_ws;
    size_t off = 0;
    auto carve = [&](size_t bytes) -> void* {
        void* p = base + off;
        off += (bytes + 255) & ~(size_t)255;
        return p;
    };
    u16*   WBF   = (u16*)  carve(WBF_TOTAL * sizeof(u16));
    float* PE    = (float*)carve((size_t)SS * DD * 4);
    float* XF    = (float*)carve((size_t)MTOK * DD * 4);          // x fp32
    float* GF    = (float*)carve((size_t)MTOK * DD * 4);          // gemm fp32 out
    u16*   XB    = (u16*)  carve((size_t)MTOK * DD * 2);          // x bf16
    u16*   BIG   = (u16*)  carve((size_t)MTOK * DFF * 2);         // emb/qkv/ff1/nbin
    u16*   ATT   = (u16*)  carve((size_t)MTOK * DD * 2);          // attn out bf16
    float* MEMF  = (float*)carve((size_t)BB * DD * 4);
    float* MEMVF = (float*)carve((size_t)BB * DD * 4);
    float* CAVEC = (float*)carve((size_t)BB * DD * 4);
    int*   NAIDX = (int*)  carve((size_t)MTOK * 4);

    auto cvt = [&](const float* s, u16* d, size_t n) {
        f32_to_bf16_kernel<<<(unsigned)((n + 255) / 256), 256, 0, stream>>>(s, d, (int)n);
    };
    auto gemm = [&](const u16* A, const u16* W, const float* bias,
                    float* Cf, u16* Cb, int M, int N, int K, int flags) {
        dim3 grid((unsigned)((N + GTN - 1) / GTN), (unsigned)(M / GTM));
        gemm_bf16_kernel<<<grid, 128, 0, stream>>>(A, W, bias, Cf, Cb, M, N, K, flags);
    };

    // ---- weight conversion (read once; then resident as bf16, fits in L2) --
    cvt(proj_w,   WBF + OFF_PROJ,  (size_t)DD * DD);
    cvt(latent_w, WBF + OFF_LATW,  (size_t)DD * LAT);
    cvt(sa_in_w,  WBF + OFF_SAIN,  (size_t)LL * 3 * DD * DD);
    cvt(sa_out_w, WBF + OFF_SAOUT, (size_t)LL * DD * DD);
    cvt(ca_kv_w,  WBF + OFF_CAKV,  (size_t)LL * 2 * DD * DD);
    cvt(ca_out_w, WBF + OFF_CAOUT, (size_t)LL * DD * DD);
    cvt(ff1_w,    WBF + OFF_FF1,   (size_t)LL * DFF * DD);
    cvt(ff2_w,    WBF + OFF_FF2,   (size_t)LL * DD * DFF);
    cvt(type_w,   WBF + OFF_TYPEW, (size_t)VV * DD);
    cvt(na_w,     WBF + OFF_NAW,   (size_t)NN * DD);
    cvt(nb_w,     WBF + OFF_NBW,   (size_t)NN * (DD + NN));
    cvt(latent,   WBF + OFF_LATIN, (size_t)BB * LAT);

    // ---- embedding + input projection + PE ---------------------------------
    pe_kernel<<<(SS * DD) / 256, 256, 0, stream>>>(PE);
    embed_kernel<<<MTOK, 128, 0, stream>>>(teacher, type_emb, na_emb, nb_emb,
                                           value_w, value_b, BIG);
    gemm(BIG, WBF + OFF_PROJ, proj_b, GF, nullptr, MTOK, DD, DD, 0);
    addpe_kernel<<<(MTOK * DD) / 256, 256, 0, stream>>>(GF, PE, XF, XB);

    // ---- memory = latent @ latent_w^T + latent_b (per batch) ---------------
    gemm(WBF + OFF_LATIN, WBF + OFF_LATW, latent_b, MEMF, WBF + OFF_MEMBF,
         BB, DD, LAT, 0);

    // ---- decoder layers ----------------------------------------------------
    for (int l = 0; l < LL; ++l) {
        gemm(XB, WBF + OFF_SAIN + (size_t)l * 3 * DD * DD, sa_in_b + l * 3 * DD,
             nullptr, BIG, MTOK, 3 * DD, DD, 0);
        attn_kernel<<<BB * HH, 64, 0, stream>>>(BIG, ATT);
        gemm(ATT, WBF + OFF_SAOUT + (size_t)l * DD * DD, sa_out_b + l * DD,
             GF, nullptr, MTOK, DD, DD, 0);
        ln_kernel<<<MTOK, 256, 0, stream>>>(XF, GF, 0, ln1_g + l * DD, ln1_b + l * DD, XB);

        // cross-attention collapsed: softmax over single memory key == 1,
        // so contribution = (mem @ Wv^T + bv) @ Wo^T + bo, per batch only.
        gemm(WBF + OFF_MEMBF,
             WBF + OFF_CAKV + (size_t)l * 2 * DD * DD + (size_t)DD * DD,
             ca_kv_b + l * 2 * DD + DD, MEMVF, WBF + OFF_MEMVBF, BB, DD, DD, 0);
        gemm(WBF + OFF_MEMVBF, WBF + OFF_CAOUT + (size_t)l * DD * DD,
             ca_out_b + l * DD, CAVEC, nullptr, BB, DD, DD, 0);
        ln_kernel<<<MTOK, 256, 0, stream>>>(XF, CAVEC, 1, ln2_g + l * DD, ln2_b + l * DD, XB);

        gemm(XB, WBF + OFF_FF1 + (size_t)l * DFF * DD, ff1_b + l * DFF,
             nullptr, BIG, MTOK, DFF, DD, 1 /*relu*/);
        gemm(BIG, WBF + OFF_FF2 + (size_t)l * DD * DFF, ff2_b + l * DD,
             GF, nullptr, MTOK, DD, DFF, 0);
        ln_kernel<<<MTOK, 256, 0, stream>>>(XF, GF, 0, ln3_g + l * DD, ln3_b + l * DD, XB);
    }

    // ---- heads + sampling --------------------------------------------------
    float* out = (float*)d_out;
    float* oT  = out;                                 // [16384,16]
    float* oNA = out + (size_t)MTOK * VV;             // [16384,64]
    float* oNB = oNA + (size_t)MTOK * NN;             // [16384,64]
    float* oV  = oNB + (size_t)MTOK * NN;             // [16384,1]
    float* oS  = oV  + (size_t)MTOK;                  // [16384,4]

    gemm(XB, WBF + OFF_TYPEW, type_b, oT,  nullptr, MTOK, VV, DD, 0);
    gemm(XB, WBF + OFF_NAW,   na_b,   oNA, nullptr, MTOK, NN, DD, 0);
    valhead_kernel<<<MTOK, 128, 0, stream>>>(XF, val_w, val_b, oV);
    sample_a_kernel<<<MTOK, 64, 0, stream>>>(oNA, gumbel_a, XB, BIG, NAIDX);
    gemm(BIG, WBF + OFF_NBW, nb_b, oNB, nullptr, MTOK, NN, DD + NN, 0);
    finalize_kernel<<<MTOK / 256, 256, 0, stream>>>(oT, oNB, gumbel_b, NAIDX, oV, oS);
}
